// SupCR_49778670961293
// MI455X (gfx1250) — matrix operand order repaired and checked by reading, not compiled, not run on hardware
//
#include <hip/hip_runtime.h>
#include <hip/hip_bf16.h>
#include <math.h>

// ---------------------------------------------------------------------------
// SupCR loss on MI455X (gfx1250, wave32, WMMA)
//   emb [4096,768] f32, labels [4096,2] f32 -> out [2] f32
// Pipeline:
//   1) normalize rows -> f16 matrix Nh (6 MB, L2-resident)
//   2) sims = (Nh @ Nh^T)/T via v_wmma_f32_16x16x32_f16, one 16x16 tile/wave
//   3) per (row, label-dim): bitonic sort dist/exp pairs in LDS, suffix scan,
//      binary search per element, reduce, atomicAdd into out[dim]
// ---------------------------------------------------------------------------

typedef __attribute__((ext_vector_type(16))) _Float16 v16h;
typedef __attribute__((ext_vector_type(8)))  _Float16 v8h;
typedef __attribute__((ext_vector_type(8)))  float    v8f;

#define M_ROWS 4096
#define K_DIM  768
#define INV_T  10.0f          // 1 / TEMPERATURE
#define EPS_V  1e-7f

// ---------------------------------------------------------------- zero output
__global__ void supcr_zero(float* out, int n) {
    int t = blockIdx.x * blockDim.x + threadIdx.x;
    if (t < n) out[t] = 0.0f;
}

// --------------------------------------------------- normalize + f16 convert
__global__ void supcr_normalize(const float* __restrict__ emb,
                                _Float16* __restrict__ nh) {
    const int row = blockIdx.x;
    const int tid = threadIdx.x;                 // 256 threads
    __shared__ float red[256];

    const float* r = emb + (size_t)row * K_DIM;
    float ss = 0.0f;
    for (int c = tid; c < K_DIM; c += 256) { float v = r[c]; ss += v * v; }
    red[tid] = ss;
    __syncthreads();
    for (int s = 128; s > 0; s >>= 1) {
        if (tid < s) red[tid] += red[tid + s];
        __syncthreads();
    }
    const float inv = 1.0f / fmaxf(sqrtf(red[0]), 1e-12f);
    _Float16* o = nh + (size_t)row * K_DIM;
    for (int c = tid; c < K_DIM; c += 256) o[c] = (_Float16)(r[c] * inv);
}

// ------------------------------------------------------- sims = Nh @ Nh^T / T
// one wave computes one 16x16 tile; K=768 in 24 steps of 32
__global__ void supcr_gemm(const _Float16* __restrict__ nh,
                           float* __restrict__ sims) {
    const int wave  = (blockIdx.x * blockDim.x + threadIdx.x) >> 5;
    const int lane  = threadIdx.x & 31;
    const int tileI = wave >> 8;                 // 256 tiles per side
    const int tileJ = wave & 255;
    const int m     = lane & 15;
    const int h     = lane >> 4;                 // half-wave selector

    // A fragment rows (tileI), B fragment rows (tileJ) — same buffer, B needs
    // column n == m of sims == row n of Nh (contiguous K per lane).
    const _Float16* aRow = nh + (size_t)(tileI * 16 + m) * K_DIM;
    const _Float16* bRow = nh + (size_t)(tileJ * 16 + m) * K_DIM;

    v8f c = {};
    for (int k0 = 0; k0 < K_DIM; k0 += 32) {
        // A k order used here (same as B): lane m holds K = h*16 .. h*16+15 contiguous
        v16h a = *(const v16h*)(aRow + k0 + h * 16);
        // B k order used here (same as A): lane n holds K = h*16 .. h*16+15 contiguous
        v16h b = *(const v16h*)(bRow + k0 + h * 16);

        c = __builtin_amdgcn_wmma_f32_16x16x32_f16(
                /*neg_a=*/false, a, /*neg_b=*/false, b,
                /*c_mod=*/(short)0, c, /*reuse_a=*/false, /*reuse_b=*/false);
    }

    // C/D layout: VGPR r -> row h*8 + r, column = lane & 15
    float* dst = sims + (size_t)(tileI * 16 + h * 8) * M_ROWS + tileJ * 16 + m;
#pragma unroll
    for (int r = 0; r < 8; ++r) dst[(size_t)r * M_ROWS] = c[r] * INV_T;
}

// ----------------------------------------------------------- per-row loss
// one 512-thread block per (row i, label-dim d)
__global__ void __launch_bounds__(512)
supcr_loss(const float* __restrict__ sims,
           const float* __restrict__ labels,
           float* __restrict__ out) {
    const int i   = blockIdx.x;
    const int d   = blockIdx.y;
    const int tid = threadIdx.x;
    const int NT  = 512;

    __shared__ float sd[M_ROWS];   // sorted distances
    __shared__ float sv[M_ROWS];   // exp(sims) -> suffix sums
    __shared__ float sl[M_ROWS];   // labels row (this dim)
    __shared__ float red[512];

    const float labi = labels[(size_t)i * 2 + d];
    const float* srow = sims + (size_t)i * M_ROWS;

    for (int j = tid; j < M_ROWS; j += NT) {
        float lj = labels[(size_t)j * 2 + d];
        sl[j] = lj;
        sd[j] = fabsf(lj - labi);
        sv[j] = __expf(srow[j]);
    }
    __syncthreads();

    // ---- bitonic sort ascending by sd, carrying sv -------------------------
    for (int k = 2; k <= M_ROWS; k <<= 1) {
        for (int jj = k >> 1; jj > 0; jj >>= 1) {
            for (int t = tid; t < M_ROWS; t += NT) {
                int ixj = t ^ jj;
                if (ixj > t) {
                    bool up = ((t & k) == 0);
                    float d0 = sd[t], d1 = sd[ixj];
                    if ((d0 > d1) == up) {
                        sd[t] = d1; sd[ixj] = d0;
                        float v0 = sv[t]; sv[t] = sv[ixj]; sv[ixj] = v0;
                    }
                }
            }
            __syncthreads();
        }
    }

    // ---- inclusive suffix scan of sv (Hillis–Steele, register staged) ------
    for (int off = 1; off < M_ROWS; off <<= 1) {
        float tmp[M_ROWS / 512];
#pragma unroll
        for (int e = 0; e < M_ROWS / 512; ++e) {
            int t = tid + e * NT;
            float nv = sv[t];
            int u = t + off;
            if (u < M_ROWS) nv += sv[u];
            tmp[e] = nv;
        }
        __syncthreads();
#pragma unroll
        for (int e = 0; e < M_ROWS / 512; ++e) sv[tid + e * NT] = tmp[e];
        __syncthreads();
    }

    // ---- per-element: lower_bound + log(denom) - sims ----------------------
    float acc = 0.0f;
    for (int j = tid; j < M_ROWS; j += NT) {
        if (j == i) continue;
        float q = fabsf(sl[j] - labi);
        int lo = 0, hi = M_ROWS;
        while (lo < hi) {
            int mid = (lo + hi) >> 1;
            if (sd[mid] < q) lo = mid + 1; else hi = mid;
        }
        float denom = (lo < M_ROWS) ? sv[lo] : 0.0f;
        acc += __logf(fmaxf(denom, EPS_V)) - srow[j];
    }

    red[tid] = acc;
    __syncthreads();
    for (int s = NT / 2; s > 0; s >>= 1) {
        if (tid < s) red[tid] += red[tid + s];
        __syncthreads();
    }
    if (tid == 0) {
        const float scale = 1.0f / ((float)M_ROWS * (float)(M_ROWS - 1));
        atomicAdd(&out[d], red[0] * scale);
    }
}

// ---------------------------------------------------------------------------
extern "C" void kernel_launch(void* const* d_in, const int* in_sizes, int n_in,
                              void* d_out, int out_size, void* d_ws, size_t ws_size,
                              hipStream_t stream) {
    const float* emb    = (const float*)d_in[0];   // [4096,768]
    const float* labels = (const float*)d_in[1];   // [4096,2]
    float* out = (float*)d_out;                    // [2]

    float*    sims = (float*)d_ws;                                  // 64 MB
    _Float16* nh   = (_Float16*)((char*)d_ws +
                                 (size_t)M_ROWS * M_ROWS * sizeof(float)); // 6 MB

    supcr_zero<<<1, 32, 0, stream>>>(out, out_size);
    supcr_normalize<<<M_ROWS, 256, 0, stream>>>(emb, nh);
    // 256x256 tiles, 8 waves (256 thr) per block -> 8192 blocks
    supcr_gemm<<<(M_ROWS / 16) * (M_ROWS / 16) / 8, 256, 0, stream>>>(nh, sims);
    supcr_loss<<<dim3(M_ROWS, 2), 512, 0, stream>>>(sims, labels, out);
}
